// GPRGNN_68341519613988
// MI455X (gfx1250) — matrix operand order repaired and checked by reading, not compiled, not run on hardware
//
#include <hip/hip_runtime.h>
#include <hip/hip_bf16.h>
#include <math.h>

typedef __attribute__((ext_vector_type(16))) __bf16 v16bf;
typedef __attribute__((ext_vector_type(2)))  __bf16 bf16x2;
typedef __attribute__((ext_vector_type(8)))  float  v8f;

#define N_NODES 100000
#define N_EDGES 3200000
#define K_HOPS  10
#define FEAT_IN 500
#define HID     64

// pack two f32 into one dword of bf16 pair.  Built as a <2 x bf16> ext_vector
// from two fptruncs so the backend fuses them into ONE v_cvt_pk_bf16_f32.
static __device__ __forceinline__ unsigned pack_bf16_pair(float lo, float hi) {
  bf16x2 t;
  t[0] = (__bf16)lo;
  t[1] = (__bf16)hi;
  return __builtin_bit_cast(unsigned, t);
}

// reinterpret 8 packed dwords as a WMMA bf16 fragment
union ABfrag {
  struct { uint4 lo; uint4 hi; } q;
  v16bf h;
};

// ---------------------------------------------------------------------------
// Fused MLP: h = relu(x@W1 + b1) @ W2 + b2   via v_wmma_f32_16x16x32_bf16.
// Block = 256 threads = 8 waves; block computes a 32x64 tile of h.
// Wave w -> (mtile = w>>2, ntile = w&3): one 16x16 WMMA output tile.
// All tiles live in LDS as packed bf16 K-pairs: every fragment register is one
// LDS dword and every fragment loads as 2x ds_load_b128, zero conversions on
// the consume side. Guards only in tail staging — EXEC all-ones at each wmma.
// ---------------------------------------------------------------------------
__global__ __launch_bounds__(256) void mlp_wmma_kernel(
    const float* __restrict__ x,  const float* __restrict__ W1,
    const float* __restrict__ b1, const float* __restrict__ W2,
    const float* __restrict__ b2, float* __restrict__ h)
{
  __shared__ unsigned xsu[32][20];     // A tile: [row][K/2] bf16 pairs (16 used)
  __shared__ unsigned wsu[64][20];     // B tile transposed: [n][K/2] (16 used)
  __shared__ unsigned h1p[32][36];     // relu(x@W1+b1) packed pairs (32 used)

  const int tid   = threadIdx.x;
  const int lane  = tid & 31;
  const int wave  = tid >> 5;
  const int half  = lane >> 4;     // 0/1 : K-group selector within fragment
  const int lr    = lane & 15;     // row (A) / col (B) within 16-tile
  const int mtile = wave >> 2;     // 0..1
  const int ntile = wave & 3;      // 0..3
  const int rowbase = blockIdx.x * 32;            // 3125 * 32 == 100000 exactly

  // staging coordinates
  const int a_r  = tid >> 3;            // 0..31 : x row within tile
  const int a_kc = (tid & 7) << 2;      // 0,4,...,28 : K offset (float4)
  const int a_kd = (tid & 7) << 1;      // dword index = a_kc/2
  const float* __restrict__ xstage = x + (size_t)(rowbase + a_r) * FEAT_IN + a_kc;
  const int b_n   = tid & 63;           // B staging: column n
  const int b_kpb = (tid >> 6) << 2;    // B staging: kpair base 0,4,8,12
  const int ncol  = ntile * 16 + lr;

  v8f acc = {};

  // ---------------- main K loop: 15 iterations, guard-free -----------------
  for (int k0 = 0; k0 < 480; k0 += 32) {
    float4 av = *reinterpret_cast<const float4*>(xstage + k0);
    unsigned pw[4];
    #pragma unroll
    for (int q = 0; q < 4; ++q) {       // scalar loads in pack order: no shuffles
      int k = k0 + ((b_kpb + q) << 1);
      float w0 = W1[(size_t)k * HID + b_n];
      float w1 = W1[(size_t)(k + 1) * HID + b_n];
      pw[q] = pack_bf16_pair(w0, w1);
    }
    __builtin_prefetch(xstage + k0 + 32, 0, 3);     // prefetch next x tile near

    unsigned pa0 = pack_bf16_pair(av.x, av.y);
    unsigned pa1 = pack_bf16_pair(av.z, av.w);

    __syncthreads();                                // prior compute done with LDS
    xsu[a_r][a_kd]     = pa0;
    xsu[a_r][a_kd + 1] = pa1;
    *reinterpret_cast<uint4*>(&wsu[b_n][b_kpb]) = make_uint4(pw[0], pw[1], pw[2], pw[3]);
    __syncthreads();

    ABfrag a, b;
    const unsigned* xr = &xsu[mtile * 16 + lr][half << 2];
    a.q.lo = *reinterpret_cast<const uint4*>(xr);        // ds_load_b128
    a.q.hi = *reinterpret_cast<const uint4*>(xr + 8);    // ds_load_b128
    const unsigned* br = &wsu[ncol][half << 2];
    b.q.lo = *reinterpret_cast<const uint4*>(br);        // ds_load_b128
    b.q.hi = *reinterpret_cast<const uint4*>(br + 8);    // ds_load_b128
    acc = __builtin_amdgcn_wmma_f32_16x16x32_bf16(
        false, a.h, false, b.h, (short)0, acc, false, false);
  }

  // ---------------- K tail: K = 480..499, zero-padded to 32 ---------------
  {
    const int k0 = 480;
    float4 av = make_float4(0.f, 0.f, 0.f, 0.f);
    if (a_kc <= 16)                                 // 480+16+3 = 499 < 500
      av = *reinterpret_cast<const float4*>(xstage + k0);
    unsigned pw[4];
    #pragma unroll
    for (int q = 0; q < 4; ++q) {
      int kp = b_kpb + q;
      float w0 = 0.f, w1 = 0.f;
      if (kp < 10) {                                // k = 498,499 max in-bounds
        int k = k0 + (kp << 1);
        w0 = W1[(size_t)k * HID + b_n];
        w1 = W1[(size_t)(k + 1) * HID + b_n];
      }
      pw[q] = pack_bf16_pair(w0, w1);
    }
    unsigned pa0 = pack_bf16_pair(av.x, av.y);
    unsigned pa1 = pack_bf16_pair(av.z, av.w);

    __syncthreads();
    xsu[a_r][a_kd]     = pa0;
    xsu[a_r][a_kd + 1] = pa1;
    *reinterpret_cast<uint4*>(&wsu[b_n][b_kpb]) = make_uint4(pw[0], pw[1], pw[2], pw[3]);
    __syncthreads();

    ABfrag a, b;
    const unsigned* xr = &xsu[mtile * 16 + lr][half << 2];
    a.q.lo = *reinterpret_cast<const uint4*>(xr);
    a.q.hi = *reinterpret_cast<const uint4*>(xr + 8);
    const unsigned* br = &wsu[ncol][half << 2];
    b.q.lo = *reinterpret_cast<const uint4*>(br);
    b.q.hi = *reinterpret_cast<const uint4*>(br + 8);
    acc = __builtin_amdgcn_wmma_f32_16x16x32_bf16(
        false, a.h, false, b.h, (short)0, acc, false, false);
  }

  // ---- bias + relu, pack even/odd column pairs via shfl, store to h1p -----
  __syncthreads();                      // all waves done reading wsu/xsu
  {
    float vals[8];
    #pragma unroll
    for (int v = 0; v < 8; ++v) {
      float val = acc[v] + b1[ncol];
      vals[v] = val > 0.f ? val : 0.f;
    }
    unsigned pk[8];
    #pragma unroll
    for (int v = 0; v < 8; ++v) {
      float nb = __shfl_xor(vals[v], 1, 32);        // neighbor column
      pk[v] = pack_bf16_pair(vals[v], nb);          // even lane: {n, n+1}
    }
    if (!(lr & 1)) {                                // closed before next wmma
      #pragma unroll
      for (int v = 0; v < 8; ++v)
        h1p[mtile * 16 + (half << 3) + v][ncol >> 1] = pk[v];
    }
  }

  // ---------------- GEMM2: K = 64, two WMMA steps, W2 staged in wsu --------
  v8f acc2 = {};
  #pragma unroll
  for (int s = 0; s < 2; ++s) {
    const int k0 = s * 32;
    unsigned pw[4];
    #pragma unroll
    for (int q = 0; q < 4; ++q) {
      int k = k0 + ((b_kpb + q) << 1);
      float w0 = W2[(size_t)k * HID + b_n];
      float w1 = W2[(size_t)(k + 1) * HID + b_n];
      pw[q] = pack_bf16_pair(w0, w1);
    }
    __syncthreads();                    // h1p visible / prior wsu reads done
    *reinterpret_cast<uint4*>(&wsu[b_n][b_kpb]) = make_uint4(pw[0], pw[1], pw[2], pw[3]);
    __syncthreads();

    ABfrag a, b;
    const unsigned* ar = &h1p[mtile * 16 + lr][(k0 >> 1) + (half << 2)];
    a.q.lo = *reinterpret_cast<const uint4*>(ar);        // ds_load_b128
    a.q.hi = *reinterpret_cast<const uint4*>(ar + 8);    // ds_load_b128
    const unsigned* br = &wsu[ncol][half << 2];
    b.q.lo = *reinterpret_cast<const uint4*>(br);
    b.q.hi = *reinterpret_cast<const uint4*>(br + 8);
    acc2 = __builtin_amdgcn_wmma_f32_16x16x32_bf16(
        false, a.h, false, b.h, (short)0, acc2, false, false);
  }
  #pragma unroll
  for (int v = 0; v < 8; ++v) {
    int m = rowbase + mtile * 16 + (half << 3) + v;
    h[(size_t)m * HID + ncol] = acc2[v] + b2[ncol];
  }
}

// ---------------------------------------------------------------------------
// Degree / normalization kernels
// ---------------------------------------------------------------------------
__global__ void deg_init_kernel(float* __restrict__ deg, int n) {
  int i = blockIdx.x * blockDim.x + threadIdx.x;
  if (i < n) deg[i] = 1.0f;               // self-loop contributes 1
}

__global__ void deg_count_kernel(const int* __restrict__ col,
                                 float* __restrict__ deg, int ne) {
  int e = blockIdx.x * blockDim.x + threadIdx.x;
  if (e < ne) atomicAdd(&deg[col[e]], 1.0f);
}

__global__ void dinv_kernel(float* __restrict__ deg, int n) {
  int i = blockIdx.x * blockDim.x + threadIdx.x;
  if (i < n) deg[i] = __frsqrt_rn(deg[i]);     // deg >= 1 always (self loop)
}

// hidden = temp[0] * h   (elementwise, float4)
__global__ void hidden_init_kernel(const float* __restrict__ h,
                                   float* __restrict__ hid,
                                   const float* __restrict__ temp, int n4) {
  int i = blockIdx.x * blockDim.x + threadIdx.x;
  if (i >= n4) return;
  float t0 = temp[0];
  float4 v = reinterpret_cast<const float4*>(h)[i];
  reinterpret_cast<float4*>(hid)[i] =
      make_float4(t0 * v.x, t0 * v.y, t0 * v.z, t0 * v.w);
}

// next[i,:] = dinv[i]^2 * cur[i,:]   (self-loop term; fully initializes next)
__global__ void hop_self_kernel(const float* __restrict__ dinv,
                                const float* __restrict__ cur,
                                float* __restrict__ nxt, int n) {
  int t = blockIdx.x * blockDim.x + threadIdx.x;   // (node, float4 group of 16)
  int i = t >> 4, g = t & 15;
  if (i >= n) return;
  float w = dinv[i] * dinv[i];
  float4 v = reinterpret_cast<const float4*>(cur + (size_t)i * HID)[g];
  reinterpret_cast<float4*>(nxt + (size_t)i * HID)[g] =
      make_float4(w * v.x, w * v.y, w * v.z, w * v.w);
}

// next[col,:] += dinv[row]*dinv[col] * cur[row,:] — block = 4 edges x 64 chans
__global__ __launch_bounds__(256) void edge_scatter_kernel(
    const int* __restrict__ row, const int* __restrict__ col,
    const float* __restrict__ dinv, const float* __restrict__ cur,
    float* __restrict__ nxt, int ne) {
  int e = blockIdx.x * 4 + (threadIdx.x >> 6);
  int c = threadIdx.x & 63;
  if (e >= ne) return;
  int r = row[e], d = col[e];
  float w = dinv[r] * dinv[d];
  atomicAdd(&nxt[(size_t)d * HID + c], w * cur[(size_t)r * HID + c]);
}

// hidden += temp[k] * next    (elementwise, float4)
__global__ void axpy_temp_kernel(float* __restrict__ hid,
                                 const float* __restrict__ nxt,
                                 const float* __restrict__ temp, int k, int n4) {
  int i = blockIdx.x * blockDim.x + threadIdx.x;
  if (i >= n4) return;
  float tk = temp[k];
  float4 a = reinterpret_cast<const float4*>(nxt)[i];
  float4 o = reinterpret_cast<float4*>(hid)[i];
  o.x += tk * a.x; o.y += tk * a.y; o.z += tk * a.z; o.w += tk * a.w;
  reinterpret_cast<float4*>(hid)[i] = o;
}

// Row-wise log_softmax over 64 cols: one wave32 per row, 2 elements per lane.
__global__ __launch_bounds__(256) void log_softmax_kernel(
    const float* __restrict__ hid, float* __restrict__ out, int n) {
  int lane = threadIdx.x & 31;
  int r = blockIdx.x * (blockDim.x >> 5) + (threadIdx.x >> 5);
  if (r >= n) return;
  const float* hr = hid + (size_t)r * HID;
  float a = hr[lane], b = hr[lane + 32];
  float m = fmaxf(a, b);
  #pragma unroll
  for (int off = 16; off > 0; off >>= 1) m = fmaxf(m, __shfl_xor(m, off, 32));
  float s = __expf(a - m) + __expf(b - m);
  #pragma unroll
  for (int off = 16; off > 0; off >>= 1) s += __shfl_xor(s, off, 32);
  float ls = __logf(s);
  out[(size_t)r * HID + lane]      = a - m - ls;
  out[(size_t)r * HID + lane + 32] = b - m - ls;
}

// ---------------------------------------------------------------------------
extern "C" void kernel_launch(void* const* d_in, const int* in_sizes, int n_in,
                              void* d_out, int out_size, void* d_ws, size_t ws_size,
                              hipStream_t stream) {
  const float* x    = (const float*)d_in[0];
  const int*   ei   = (const int*)  d_in[1];
  const float* W1   = (const float*)d_in[2];
  const float* b1   = (const float*)d_in[3];
  const float* W2   = (const float*)d_in[4];
  const float* b2   = (const float*)d_in[5];
  const float* temp = (const float*)d_in[6];
  const int* row = ei;               // edge_index[0] : source
  const int* col = ei + N_EDGES;     // edge_index[1] : target
  float* out = (float*)d_out;

  // Workspace carve-up (float-aligned): h | next | hidden | deg(->dinv)
  float* h   = (float*)d_ws;
  float* nxt = h   + (size_t)N_NODES * HID;
  float* hid = nxt + (size_t)N_NODES * HID;
  float* deg = hid + (size_t)N_NODES * HID;    // becomes dinv in place

  const int NELT4 = (N_NODES * HID) / 4;       // 1.6M float4

  // 1) MLP via WMMA
  mlp_wmma_kernel<<<N_NODES / 32, 256, 0, stream>>>(x, W1, b1, W2, b2, h);

  // 2) gcn_norm
  deg_init_kernel <<<(N_NODES + 255) / 256, 256, 0, stream>>>(deg, N_NODES);
  deg_count_kernel<<<(N_EDGES + 255) / 256, 256, 0, stream>>>(col, deg, N_EDGES);
  dinv_kernel     <<<(N_NODES + 255) / 256, 256, 0, stream>>>(deg, N_NODES);

  // 3) hidden = temp[0] * h ; cur = h
  hidden_init_kernel<<<(NELT4 + 255) / 256, 256, 0, stream>>>(h, hid, temp, NELT4);

  float* cur = h;
  float* nx  = nxt;
  for (int k = 0; k < K_HOPS; ++k) {
    hop_self_kernel<<<((N_NODES * 16) + 255) / 256, 256, 0, stream>>>(deg, cur, nx, N_NODES);
    edge_scatter_kernel<<<N_EDGES / 4, 256, 0, stream>>>(row, col, deg, cur, nx, N_EDGES);
    axpy_temp_kernel<<<(NELT4 + 255) / 256, 256, 0, stream>>>(hid, nx, temp, k + 1, NELT4);
    float* t = cur; cur = nx; nx = t;          // deterministic pointer swap
  }

  // 4) log_softmax -> d_out
  log_softmax_kernel<<<(N_NODES + 7) / 8, 256, 0, stream>>>(hid, out, N_NODES);
}
